// EctLayer_38285338476717
// MI455X (gfx1250) — matrix operand order repaired and checked by reading, not compiled, not run on hardware
//
#include <hip/hip_runtime.h>

typedef __attribute__((ext_vector_type(16))) _Float16 v16h;
typedef __attribute__((ext_vector_type(8)))  float    v8f;

#define T_THETA   32
#define NSTEPS    32
#define TILE      32
#define NODES_PB  512   // nodes per block chunk (16 tiles/round-loop per block)

// sigmoid(500*(lin-nh)) = 0.5*tanh(250*(lin-nh)) + 0.5   (1 TRANS op instead of 2)
#define KHALF 250.0f

__global__ void zero_out_kernel(float* __restrict__ out, int n) {
    int i = blockIdx.x * blockDim.x + threadIdx.x;
    if (i < n) out[i] = 0.0f;
}

__launch_bounds__(256)
__global__ void ect_wmma_kernel(const float* __restrict__ x,     // [N,3]
                                const float* __restrict__ v,     // [3,32]
                                const float* __restrict__ lin,   // [32]
                                const int*   __restrict__ index, // [N] sorted
                                const int*   __restrict__ nsegp, // [1]
                                float*       __restrict__ out,   // [S,32,32]
                                int n) {
    __shared__ float vs[3 * T_THETA];      // v cached
    __shared__ float lsc[NSTEPS];          // lin[s] * 250
    __shared__ float xs[TILE * 3];         // x tile
    __shared__ int   segs[TILE];           // segment ids of tile (sorted)
    __shared__ float nh[TILE * T_THETA];   // nh[node][theta]

    const int tid    = threadIdx.x;
    const int lane   = tid & 31;
    const int wave   = tid >> 5;           // wave handles steps 4w..4w+3
    const int numSeg = nsegp[0];

    if (tid < 3 * T_THETA) vs[tid]  = v[tid];
    if (tid < NSTEPS)      lsc[tid] = lin[tid] * KHALF;

    // persistent WMMA accumulators: [stepLocal 0..3][thetaHalf 0..1] = 64 VGPRs
    v8f acc[4][2];
#pragma unroll
    for (int sl = 0; sl < 4; ++sl)
#pragma unroll
        for (int hf = 0; hf < 2; ++hf) acc[sl][hf] = (v8f)0.0f;

    int segBase = -1000000;   // forces window init on first tile

    // flush accumulators to global output with float atomics
    auto flushAcc = [&]() {
#pragma unroll
        for (int sl = 0; sl < 4; ++sl) {
            const int s = wave * 4 + sl;
#pragma unroll
            for (int hf = 0; hf < 2; ++hf) {
#pragma unroll
                for (int r = 0; r < 8; ++r) {
                    // D layout: VGPR r, lanes 0-15: M=r,N=lane; lanes 16-31: M=r+8,N=lane-16
                    float val = acc[sl][hf][r];
                    int M    = r + ((lane >= 16) ? 8 : 0);
                    int Ncol = lane & 15;
                    int sg   = segBase + M;
                    if (val != 0.0f && sg >= 0 && sg < numSeg)
                        atomicAdd(&out[sg * (NSTEPS * T_THETA) + s * T_THETA + hf * 16 + Ncol], val);
                    acc[sl][hf][r] = 0.0f;
                }
            }
        }
    };

    const int chunkBase = blockIdx.x * NODES_PB;
    const int chunkEnd  = (chunkBase + NODES_PB < n) ? (chunkBase + NODES_PB) : n;

    for (int tb = chunkBase; tb < chunkEnd; tb += TILE) {
        __syncthreads();   // previous round's LDS consumers done
        if (tid < TILE * 3) xs[tid]  = x[tb * 3 + tid];
        if (tid < TILE)     segs[tid] = index[tb + tid];
        __syncthreads();

        // nh[node][theta] = x[node,:] . v[:,theta]   (1024 entries, 4 per thread)
#pragma unroll
        for (int i = 0; i < 4; ++i) {
            int e = tid + 256 * i;
            int node = e >> 5, th = e & 31;
            nh[e] = xs[node * 3 + 0] * vs[th]
                  + xs[node * 3 + 1] * vs[32 + th]
                  + xs[node * 3 + 2] * vs[64 + th];
        }
        __syncthreads();

        // index is sorted -> min/max of tile are endpoints (uniform across block)
        int minSeg = segs[0];
        int maxSeg = segs[TILE - 1];
        if (maxSeg >= segBase + 16) {   // window shift (rare: ~1562 nodes/segment)
            flushAcc();
            segBase = minSeg;
        }

        // Build one-hot A [16 segs x 32 nodes] in f16 A-operand layout:
        // lanes 0-15 (M=lane):    q<8 -> K=q,   q>=8 -> K=q+8
        // lanes 16-31 (M=lane-16):q<8 -> K=q+8, q>=8 -> K=q+16
        v16h a;
        {
            const int  Mrow = lane & 15;
            const bool hiL  = lane >= 16;
#pragma unroll
            for (int q = 0; q < 16; ++q) {
                int k = hiL ? ((q < 8) ? q + 8 : q + 16)
                            : ((q < 8) ? q     : q + 8);
                a[q] = (_Float16)((segs[k] - segBase == Mrow) ? 1.0f : 0.0f);
            }
        }

        // Each wave: its 4 steps x 2 theta-halves; B built directly in operand layout
#pragma unroll
        for (int sl = 0; sl < 4; ++sl) {
            const int   s  = wave * 4 + sl;
            const float ls = lsc[s];
#pragma unroll
            for (int hf = 0; hf < 2; ++hf) {
                const int th    = hf * 16 + (lane & 15);
                const int kbase = (lane < 16) ? 0 : 8;    // B: lanes 0-15 K=0-7,16-23; 16-31 K=8-15,24-31
                v16h b;
#pragma unroll
                for (int q = 0; q < 16; ++q) {
                    int   k  = kbase + q + ((q >= 8) ? 8 : 0);
                    float z  = ls - nh[k * 32 + th] * KHALF;        // 250*(lin[s]-nh)
                    float t  = __builtin_amdgcn_tanhf(z);           // v_tanh_f32 (CDNA5)
                    float sg = __builtin_fmaf(t, 0.5f, 0.5f);       // sigmoid(500*(lin-nh))
                    b[q] = (_Float16)sg;                            // in [0,1]
                }
                acc[sl][hf] = __builtin_amdgcn_wmma_f32_16x16x32_f16(
                    false, a, false, b, (short)0, acc[sl][hf], false, false);
            }
        }
    }

    flushAcc();   // final window
}

extern "C" void kernel_launch(void* const* d_in, const int* in_sizes, int n_in,
                              void* d_out, int out_size, void* d_ws, size_t ws_size,
                              hipStream_t stream) {
    (void)n_in; (void)d_ws; (void)ws_size;
    const float* x     = (const float*)d_in[0];   // [N,3] f32
    const float* v     = (const float*)d_in[1];   // [3,32] f32
    const float* lin   = (const float*)d_in[2];   // [32] f32
    const int*   index = (const int*)d_in[3];     // [N] i32 (sorted)
    const int*   nseg  = (const int*)d_in[4];     // [1] i32
    float*       out   = (float*)d_out;           // [S,32,32] f32

    const int n = in_sizes[0] / 3;                // 200000 nodes

    zero_out_kernel<<<(out_size + 255) / 256, 256, 0, stream>>>(out, out_size);

    const int blocks = (n + NODES_PB - 1) / NODES_PB;
    ect_wmma_kernel<<<blocks, 256, 0, stream>>>(x, v, lin, index, nseg, out, n);
}
